// PartialFormerBlock_37984690766447
// MI455X (gfx1250) — matrix operand order repaired and checked by reading, not compiled, not run on hardware
//
#include <hip/hip_runtime.h>

#define LTOK 3136
#define DIM  384
#define HEADS 6
#define HD   64
#define HEXP 24
#define PRd  96
#define HID  1536
#define NP   784
#define NF   392
#define NBT  1568
#define MQ   1569
#define MQP  1600
#define HS   56
#define KCH  32
#define SCALE_F 0.125f

typedef __attribute__((ext_vector_type(16))) _Float16 v16h;
typedef __attribute__((ext_vector_type(8)))  _Float16 v8h;
typedef __attribute__((ext_vector_type(8)))  float    v8f;

__device__ __forceinline__ v16h load_a_frag(const _Float16* p) {
    v8h lo = *(const v8h*)p;
    v8h hi = *(const v8h*)(p + 8);
    v16h r;
#pragma unroll
    for (int i = 0; i < 8; ++i) { r[i] = lo[i]; r[i + 8] = hi[i]; }
    return r;
}

__device__ __forceinline__ v8f wmma16(v16h a, v16h b, v8f c) {
    return __builtin_amdgcn_wmma_f32_16x16x32_f16(false, a, false, b, (short)0, c, false, false);
}

// ---------------- depthwise 3x3 conv + bias + residual ----------------
__global__ __launch_bounds__(256) void k_conv(const float* __restrict__ x,
                                              const float* __restrict__ w,
                                              const float* __restrict__ b,
                                              float* __restrict__ c) {
    int idx = blockIdx.x * 256 + threadIdx.x;
    if (idx >= LTOK * DIM) return;
    int d = idx % DIM, r = idx / DIM;
    int i = r / HS, j = r % HS;
    float acc = x[idx] + b[d];
#pragma unroll
    for (int ki = 0; ki < 3; ++ki) {
        int ii = i + ki - 1;
        if (ii < 0 || ii >= HS) continue;
#pragma unroll
        for (int kj = 0; kj < 3; ++kj) {
            int jj = j + kj - 1;
            if (jj < 0 || jj >= HS) continue;
            acc += w[d * 9 + ki * 3 + kj] * x[(ii * HS + jj) * DIM + d];
        }
    }
    c[idx] = acc;
}

// ---------------- per-patch means ----------------
__global__ __launch_bounds__(128) void k_means(const float* __restrict__ c,
                                               float* __restrict__ means) {
    int p = blockIdx.x, tid = threadIdx.x;
    int ip = p / 28, jp = p % 28;
    float s = 0.f;
    for (int e = tid; e < 4 * DIM; e += 128) {
        int t = e / DIM, d = e % DIM;
        int row = (2 * ip + (t >> 1)) * HS + 2 * jp + (t & 1);
        s += c[row * DIM + d];
    }
    __shared__ float red[128];
    red[tid] = s; __syncthreads();
    for (int st = 64; st > 0; st >>= 1) {
        if (tid < st) red[tid] += red[tid + st];
        __syncthreads();
    }
    if (tid == 0) means[p] = red[0] / (4.f * DIM);
}

// ---------------- stable descending rank (argsort(-means)) ----------------
__global__ __launch_bounds__(256) void k_rank(const float* __restrict__ means,
                                              int* __restrict__ rankA,
                                              int* __restrict__ order) {
    int tid = threadIdx.x;
    for (int i = tid; i < NP; i += 256) {
        float mi = means[i];
        int rk = 0;
        for (int j = 0; j < NP; ++j) {
            float mj = means[j];
            if (mj > mi || (mj == mi && j < i)) rk++;
        }
        rankA[i] = rk;
        order[rk] = i;
    }
}

// ---------------- gather fg / bg token matrices (f16, padded to MQP rows each) ----------------
__global__ __launch_bounds__(256) void k_gather(const float* __restrict__ c,
                                                const float* __restrict__ qa,
                                                const int* __restrict__ order,
                                                _Float16* __restrict__ xf,
                                                _Float16* __restrict__ xb) {
    int idx = blockIdx.x * 256 + threadIdx.x;
    if (idx >= 2 * MQP * DIM) return;
    int row = idx / DIM, d = idx % DIM;
    if (row < MQP) {
        float v;
        if (row == 0) v = qa[d];
        else if (row < MQ) {
            int tok = row - 1, rk = tok >> 2, t = tok & 3;
            int p = order[rk], ip = p / 28, jp = p % 28;
            int srow = (2 * ip + (t >> 1)) * HS + 2 * jp + (t & 1);
            v = c[srow * DIM + d];
        } else v = 0.f;
        xf[(size_t)row * DIM + d] = (_Float16)v;
    } else {
        int tok = row - MQP;
        float v = 0.f;
        if (tok < NBT) {
            int rk = NF + (tok >> 2), t = tok & 3;
            int p = order[rk], ip = p / 28, jp = p % 28;
            int srow = (2 * ip + (t >> 1)) * HS + 2 * jp + (t & 1);
            v = c[srow * DIM + d];
        }
        xb[(size_t)tok * DIM + d] = (_Float16)v;
    }
}

// ---------------- f32 -> f16 ----------------
__global__ __launch_bounds__(256) void k_f2h(const float* __restrict__ s,
                                             _Float16* __restrict__ d, int n) {
    int i = blockIdx.x * 256 + threadIdx.x;
    if (i < n) d[i] = (_Float16)s[i];
}

// ---------------- WMMA GEMM: C = A(MxK) @ W(NxK)^T + bias ----------------
// 128-thread block = 2x2 waves, 64x64 block tile, 32x32 per wave (4 WMMAs / k-step).
// mode 0: outF = acc+bias ; mode 2: outH = silu(acc+bias) ; mode 3: outF = res+acc+bias
__device__ __forceinline__ void gemm_store(v8f acc, int row0, int col,
                                           const float* __restrict__ bias,
                                           const float* __restrict__ res,
                                           float* __restrict__ outF,
                                           _Float16* __restrict__ outH,
                                           int N, int mode) {
    float bn = bias[col];
#pragma unroll
    for (int r = 0; r < 8; ++r) {
        size_t row = (size_t)row0 + r;
        float v = acc[r] + bn;
        if (mode == 2) {
            float sg = v / (1.f + __expf(-v));
            outH[row * N + col] = (_Float16)sg;
        } else if (mode == 3) {
            outF[row * N + col] = v + res[row * N + col];
        } else {
            outF[row * N + col] = v;
        }
    }
}

__global__ __launch_bounds__(128) void k_gemm(const _Float16* __restrict__ A,
                                              const _Float16* __restrict__ W,
                                              const float* __restrict__ bias,
                                              const float* __restrict__ res,
                                              float* __restrict__ outF,
                                              _Float16* __restrict__ outH,
                                              int M, int N, int K, int mode) {
    int wave = threadIdx.x >> 5;
    int lane = threadIdx.x & 31;
    int ln = lane & 15, hg = lane >> 4;
    int wm = wave >> 1, wn = wave & 1;
    int m0 = blockIdx.y * 64 + wm * 32;
    int n0 = blockIdx.x * 64 + wn * 32;
    const _Float16* a0 = A + (size_t)(m0 + ln) * K + hg * 16;
    const _Float16* a1 = a0 + (size_t)16 * K;
    const _Float16* b0 = W + (size_t)(n0 + ln) * K + hg * 16;
    const _Float16* b1 = b0 + (size_t)16 * K;
    v8f acc00 = {}, acc01 = {}, acc10 = {}, acc11 = {};
#pragma unroll 2
    for (int k0 = 0; k0 < K; k0 += 32) {
        v16h af0 = load_a_frag(a0 + k0);
        v16h af1 = load_a_frag(a1 + k0);
        v16h bf0 = *(const v16h*)(b0 + k0);
        v16h bf1 = *(const v16h*)(b1 + k0);
        acc00 = wmma16(af0, bf0, acc00);
        acc01 = wmma16(af0, bf1, acc01);
        acc10 = wmma16(af1, bf0, acc10);
        acc11 = wmma16(af1, bf1, acc11);
    }
    int rbase = m0 + hg * 8;
    gemm_store(acc00, rbase,      n0 + ln,      bias, res, outF, outH, N, mode);
    gemm_store(acc01, rbase,      n0 + 16 + ln, bias, res, outF, outH, N, mode);
    gemm_store(acc10, rbase + 16, n0 + ln,      bias, res, outF, outH, N, mode);
    gemm_store(acc11, rbase + 16, n0 + 16 + ln, bias, res, outF, outH, N, mode);
}

// ---------------- split qkv into q16 / k16 / vT16 ----------------
__global__ __launch_bounds__(256) void k_qkvpost(const float* __restrict__ qkv,
                                                 _Float16* __restrict__ q16,
                                                 _Float16* __restrict__ k16,
                                                 _Float16* __restrict__ vT) {
    int idx = blockIdx.x * 256 + threadIdx.x;
    if (idx >= MQP * DIM) return;
    int row = idx / DIM, d = idx % DIM;
    const float* rp = qkv + (size_t)row * (3 * DIM);
    q16[idx] = (_Float16)rp[d];
    k16[idx] = (_Float16)rp[DIM + d];
    vT[(size_t)d * MQP + row] = (_Float16)rp[2 * DIM + d];
}

// ---------------- MMSA: flash-style two-pass, 6 waves = 6 heads ----------------
__global__ __launch_bounds__(192) void k_attn(const _Float16* __restrict__ q16,
                                              const _Float16* __restrict__ k16,
                                              const _Float16* __restrict__ vT,
                                              const float* __restrict__ ap1w,
                                              const float* __restrict__ ap1b,
                                              const float* __restrict__ ap2w,
                                              const float* __restrict__ ap2b,
                                              _Float16* __restrict__ outH) {
    __shared__ float S[HEADS][16][KCH + 1];
    __shared__ _Float16 A2s[HEADS][16][KCH + 2];
    __shared__ float mrow[16][HEXP + 1];
    __shared__ float srow[16][HEXP + 1];
    __shared__ float w1s[HEXP][HEADS], w2s[HEADS][HEXP];
    __shared__ float b1s[HEXP], b2s[HEADS];

    int tid = threadIdx.x;
    int h = tid >> 5, lane = tid & 31, ln = lane & 15, hg = lane >> 4;
    if (tid < HEXP * HEADS) {
        w1s[tid / HEADS][tid % HEADS] = ap1w[tid];
        w2s[tid / HEXP][tid % HEXP] = ap2w[tid];
    }
    if (tid < HEXP) b1s[tid] = ap1b[tid];
    if (tid < HEADS) b2s[tid] = ap2b[tid];
    __syncthreads();

    int m0 = blockIdx.x * 16;
    const _Float16* qrow = q16 + (size_t)(m0 + ln) * DIM + h * HD + hg * 16;
    v16h qf0 = load_a_frag(qrow);
    v16h qf1 = load_a_frag(qrow + 32);

    float mst[2] = {-3.0e38f, -3.0e38f};
    float sst[2] = {0.f, 0.f};

    // -------- pass 1: online softmax stats per (q, channel) --------
    for (int k0 = 0; k0 < MQP; k0 += KCH) {
#pragma unroll
        for (int sub = 0; sub < 2; ++sub) {
            const _Float16* kr = k16 + (size_t)(k0 + sub * 16 + ln) * DIM + h * HD + hg * 16;
            v16h bf0 = *(const v16h*)kr;
            v16h bf1 = *(const v16h*)(kr + 32);
            v8f a = {};
            a = wmma16(qf0, bf0, a);
            a = wmma16(qf1, bf1, a);
#pragma unroll
            for (int r = 0; r < 8; ++r) S[h][hg * 8 + r][sub * 16 + ln] = a[r] * SCALE_F;
        }
        __syncthreads();
#pragma unroll
        for (int pp = 0; pp < 2; ++pp) {
            int idx = tid + pp * 192;
            int qq = idx / HEXP, cc = idx - qq * HEXP;
            float m = mst[pp], s = sst[pp];
            for (int kk = 0; kk < KCH; ++kk) {
                if (k0 + kk >= MQ) break;
                float a1 = b1s[cc];
#pragma unroll
                for (int hh = 0; hh < HEADS; ++hh) a1 += S[hh][qq][kk] * w1s[cc][hh];
                if (a1 > m) { s = s * __expf(m - a1) + 1.f; m = a1; }
                else s += __expf(a1 - m);
            }
            mst[pp] = m; sst[pp] = s;
        }
        __syncthreads();
    }
#pragma unroll
    for (int pp = 0; pp < 2; ++pp) {
        int idx = tid + pp * 192;
        int qq = idx / HEXP, cc = idx - qq * HEXP;
        mrow[qq][cc] = mst[pp];
        srow[qq][cc] = sst[pp];
    }
    __syncthreads();

    // -------- pass 2: attention weights + A2 @ V via WMMA --------
    v8f o0 = {}, o1 = {}, o2 = {}, o3 = {};
    for (int k0 = 0; k0 < MQP; k0 += KCH) {
#pragma unroll
        for (int sub = 0; sub < 2; ++sub) {
            const _Float16* kr = k16 + (size_t)(k0 + sub * 16 + ln) * DIM + h * HD + hg * 16;
            v16h bf0 = *(const v16h*)kr;
            v16h bf1 = *(const v16h*)(kr + 32);
            v8f a = {};
            a = wmma16(qf0, bf0, a);
            a = wmma16(qf1, bf1, a);
#pragma unroll
            for (int r = 0; r < 8; ++r) S[h][hg * 8 + r][sub * 16 + ln] = a[r] * SCALE_F;
        }
        __syncthreads();
        for (int pos = tid; pos < 16 * KCH; pos += 192) {
            int qq = pos >> 5, kk = pos & 31;
            if (k0 + kk >= MQ) {
#pragma unroll
                for (int hh = 0; hh < HEADS; ++hh) A2s[hh][qq][kk] = (_Float16)0.f;
            } else {
                float a2[HEADS];
#pragma unroll
                for (int hh = 0; hh < HEADS; ++hh) a2[hh] = b2s[hh];
                for (int cc = 0; cc < HEXP; ++cc) {
                    float a1 = b1s[cc];
#pragma unroll
                    for (int hh = 0; hh < HEADS; ++hh) a1 += S[hh][qq][kk] * w1s[cc][hh];
                    float w = __expf(a1 - mrow[qq][cc]) / srow[qq][cc];
#pragma unroll
                    for (int hh = 0; hh < HEADS; ++hh) a2[hh] += w * w2s[hh][cc];
                }
#pragma unroll
                for (int hh = 0; hh < HEADS; ++hh) A2s[hh][qq][kk] = (_Float16)a2[hh];
            }
        }
        __syncthreads();
        v16h af;
        int kb = hg * 16;
#pragma unroll
        for (int e = 0; e < 16; ++e) {
            int kl = kb + e;
            af[e] = A2s[h][ln][kl];
        }
        const _Float16* vb = vT + (size_t)(h * HD + ln) * MQP + k0 + hg * 16;
        o0 = wmma16(af, *(const v16h*)(vb), o0);
        o1 = wmma16(af, *(const v16h*)(vb + (size_t)16 * MQP), o1);
        o2 = wmma16(af, *(const v16h*)(vb + (size_t)32 * MQP), o2);
        o3 = wmma16(af, *(const v16h*)(vb + (size_t)48 * MQP), o3);
        __syncthreads();
    }
#pragma unroll
    for (int r = 0; r < 8; ++r) {
        size_t row = (size_t)m0 + hg * 8 + r;
        _Float16* op = outH + row * DIM + h * HD + ln;
        op[0]  = (_Float16)o0[r];
        op[16] = (_Float16)o1[r];
        op[32] = (_Float16)o2[r];
        op[48] = (_Float16)o3[r];
    }
}

// ---------------- SQA: 1-query attention + MLP, single block ----------------
__global__ __launch_bounds__(256) void k_sqa(const float* __restrict__ kv,
                                             const float* __restrict__ f_out,
                                             const float* __restrict__ sp1w,
                                             const float* __restrict__ sp1b,
                                             const float* __restrict__ sp2w,
                                             const float* __restrict__ sp2b,
                                             const float* __restrict__ pnw,
                                             const float* __restrict__ pnb,
                                             float* __restrict__ logits,
                                             float* __restrict__ addvec,
                                             float* __restrict__ qa2_out) {
    __shared__ float red[256];
    __shared__ float q[DIM];
    __shared__ float x1s[DIM];
    __shared__ float pmax[HEADS], psum[HEADS];
    __shared__ float s1[PRd], y[PRd];
    __shared__ float lnm, lnv;
    int tid = threadIdx.x;

    for (int d = tid; d < DIM; d += 256) { float v = f_out[d]; q[d] = v; qa2_out[d] = v; }
    __syncthreads();

    for (int i = tid; i < HEADS * NBT; i += 256) {
        int hh = i / NBT, t = i % NBT;
        const float* kr = kv + (size_t)t * (2 * DIM) + hh * HD;
        float a = 0.f;
#pragma unroll 8
        for (int d = 0; d < HD; ++d) a += q[hh * HD + d] * kr[d];
        logits[i] = a * SCALE_F;
    }
    __syncthreads();

    for (int hh = 0; hh < HEADS; ++hh) {
        float m = -3.0e38f;
        for (int t = tid; t < NBT; t += 256) m = fmaxf(m, logits[hh * NBT + t]);
        red[tid] = m; __syncthreads();
        for (int st = 128; st > 0; st >>= 1) { if (tid < st) red[tid] = fmaxf(red[tid], red[tid + st]); __syncthreads(); }
        m = red[0]; __syncthreads();
        float s = 0.f;
        for (int t = tid; t < NBT; t += 256) s += __expf(logits[hh * NBT + t] - m);
        red[tid] = s; __syncthreads();
        for (int st = 128; st > 0; st >>= 1) { if (tid < st) red[tid] += red[tid + st]; __syncthreads(); }
        if (tid == 0) { pmax[hh] = m; psum[hh] = red[0]; }
        __syncthreads();
    }
    for (int i = tid; i < HEADS * NBT; i += 256) {
        int hh = i / NBT;
        logits[i] = __expf(logits[i] - pmax[hh]) / psum[hh];
    }
    __syncthreads();
    for (int p = tid; p < DIM; p += 256) {
        int hh = p >> 6, d = p & 63;
        float a = 0.f;
        for (int t = 0; t < NBT; ++t) a += logits[hh * NBT + t] * kv[(size_t)t * (2 * DIM) + DIM + hh * HD + d];
        x1s[p] = a;
    }
    __syncthreads();
    for (int o = tid; o < PRd; o += 256) {
        float a = sp1b[o];
        for (int d = 0; d < DIM; ++d) a += sp1w[o * DIM + d] * x1s[d];
        s1[o] = a;
    }
    __syncthreads();
    if (tid == 0) {
        float s = 0.f, ss = 0.f;
        for (int o = 0; o < PRd; ++o) { s += s1[o]; ss += s1[o] * s1[o]; }
        float mm = s / PRd;
        lnm = mm; lnv = ss / PRd - mm * mm;
    }
    __syncthreads();
    for (int o = tid; o < PRd; o += 256) {
        float v = (s1[o] - lnm) * rsqrtf(lnv + 1e-5f) * pnw[o] + pnb[o];
        y[o] = fmaxf(v, 0.f);
    }
    __syncthreads();
    for (int d = tid; d < DIM; d += 256) {
        float a = sp2b[d];
        for (int o = 0; o < PRd; ++o) a += sp2w[d * PRd + o] * y[o];
        addvec[d] = a;
    }
}

// ---------------- scatter + LN1 + residual + LN2 (writes x2 f32 + ln2 f16) ----------------
__global__ __launch_bounds__(128) void k_scatter(const float* __restrict__ x,
                                                 const float* __restrict__ c,
                                                 const float* __restrict__ f_out,
                                                 const float* __restrict__ addvec,
                                                 const int* __restrict__ rankA,
                                                 const float* __restrict__ n1w,
                                                 const float* __restrict__ n1b,
                                                 const float* __restrict__ n2w,
                                                 const float* __restrict__ n2b,
                                                 float* __restrict__ x2,
                                                 _Float16* __restrict__ xln16) {
    int r = blockIdx.x, tid = threadIdx.x;
    int i2 = r / HS, j2 = r % HS;
    int p = (i2 >> 1) * 28 + (j2 >> 1);
    int t = (i2 & 1) * 2 + (j2 & 1);
    int rk = rankA[p];
    __shared__ float buf[DIM];
    __shared__ float rs[128], rss[128];
    __shared__ float smean, svar;

    if (rk < NF) {
        int tok = rk * 4 + t;
        for (int d = tid; d < DIM; d += 128) buf[d] = f_out[(size_t)(tok + 1) * DIM + d];
    } else {
        for (int d = tid; d < DIM; d += 128) buf[d] = c[(size_t)r * DIM + d] + addvec[d];
    }
    __syncthreads();
    float s = 0.f, ss = 0.f;
    for (int d = tid; d < DIM; d += 128) { float v = buf[d]; s += v; ss += v * v; }
    rs[tid] = s; rss[tid] = ss; __syncthreads();
    for (int st = 64; st > 0; st >>= 1) {
        if (tid < st) { rs[tid] += rs[tid + st]; rss[tid] += rss[tid + st]; }
        __syncthreads();
    }
    if (tid == 0) { float mm = rs[0] / DIM; smean = mm; svar = rss[0] / DIM - mm * mm; }
    __syncthreads();
    float inv1 = rsqrtf(svar + 1e-5f), m1 = smean;
    __syncthreads();
    for (int d = tid; d < DIM; d += 128) {
        float v = x[(size_t)r * DIM + d] + (buf[d] - m1) * inv1 * n1w[d] + n1b[d];
        x2[(size_t)r * DIM + d] = v;
        buf[d] = v;                 // reuse for LN2
    }
    __syncthreads();
    s = 0.f; ss = 0.f;
    for (int d = tid; d < DIM; d += 128) { float v = buf[d]; s += v; ss += v * v; }
    rs[tid] = s; rss[tid] = ss; __syncthreads();
    for (int st = 64; st > 0; st >>= 1) {
        if (tid < st) { rs[tid] += rs[tid + st]; rss[tid] += rss[tid + st]; }
        __syncthreads();
    }
    if (tid == 0) { float mm = rs[0] / DIM; smean = mm; svar = rss[0] / DIM - mm * mm; }
    __syncthreads();
    float inv2 = rsqrtf(svar + 1e-5f), m2 = smean;
    for (int d = tid; d < DIM; d += 128) {
        float v = (buf[d] - m2) * inv2 * n2w[d] + n2b[d];
        xln16[(size_t)r * DIM + d] = (_Float16)v;
    }
}

// ======================= host launcher =======================
extern "C" void kernel_launch(void* const* d_in, const int* in_sizes, int n_in,
                              void* d_out, int out_size, void* d_ws, size_t ws_size,
                              hipStream_t stream) {
    const float* x       = (const float*)d_in[0];
    const float* qa      = (const float*)d_in[1];
    const float* conv_w  = (const float*)d_in[2];
    const float* conv_b  = (const float*)d_in[3];
    const float* n1w     = (const float*)d_in[4];
    const float* n1b     = (const float*)d_in[5];
    const float* n2w     = (const float*)d_in[6];
    const float* n2b     = (const float*)d_in[7];
    const float* qkv_w   = (const float*)d_in[8];
    const float* qkv_b   = (const float*)d_in[9];
    const float* mproj_w = (const float*)d_in[10];
    const float* mproj_b = (const float*)d_in[11];
    const float* ap1_w   = (const float*)d_in[12];
    const float* ap1_b   = (const float*)d_in[13];
    const float* ap2_w   = (const float*)d_in[14];
    const float* ap2_b   = (const float*)d_in[15];
    const float* kv_w    = (const float*)d_in[16];
    const float* kv_b    = (const float*)d_in[17];
    const float* sp1_w   = (const float*)d_in[18];
    const float* sp1_b   = (const float*)d_in[19];
    const float* sp2_w   = (const float*)d_in[20];
    const float* sp2_b   = (const float*)d_in[21];
    const float* pn_w    = (const float*)d_in[22];
    const float* pn_b    = (const float*)d_in[23];
    const float* fc1_w   = (const float*)d_in[24];
    const float* fc1_b   = (const float*)d_in[25];
    const float* fc2_w   = (const float*)d_in[26];
    const float* fc2_b   = (const float*)d_in[27];
    float* out = (float*)d_out;

    char* ws = (char*)d_ws;
    size_t off = 0;
    auto alloc = [&](size_t b) -> void* {
        void* p = ws + off;
        off = (off + b + 255) & ~(size_t)255;
        return p;
    };
    float*    c        = (float*)alloc((size_t)LTOK * DIM * 4);
    float*    means    = (float*)alloc((size_t)NP * 4);
    int*      rankA    = (int*)alloc((size_t)NP * 4);
    int*      order    = (int*)alloc((size_t)NP * 4);
    _Float16* xf16     = (_Float16*)alloc((size_t)MQP * DIM * 2);
    _Float16* xb16     = (_Float16*)alloc((size_t)MQP * DIM * 2);   // padded to MQP rows
    _Float16* qkvW16   = (_Float16*)alloc((size_t)3 * DIM * DIM * 2);
    _Float16* mprojW16 = (_Float16*)alloc((size_t)DIM * DIM * 2);
    _Float16* kvW16    = (_Float16*)alloc((size_t)2 * DIM * DIM * 2);
    _Float16* fc1W16   = (_Float16*)alloc((size_t)HID * DIM * 2);
    _Float16* fc2W16   = (_Float16*)alloc((size_t)DIM * HID * 2);
    float*    qkv_out  = (float*)alloc((size_t)MQP * 3 * DIM * 4);
    _Float16* q16      = (_Float16*)alloc((size_t)MQP * DIM * 2);
    _Float16* k16      = (_Float16*)alloc((size_t)MQP * DIM * 2);
    _Float16* vT16     = (_Float16*)alloc((size_t)DIM * MQP * 2);
    _Float16* attn16   = (_Float16*)alloc((size_t)MQP * DIM * 2);
    float*    f_out    = (float*)alloc((size_t)MQP * DIM * 4);
    float*    kv_out   = (float*)alloc((size_t)MQP * 2 * DIM * 4);  // padded to MQP rows
    float*    logits   = (float*)alloc((size_t)HEADS * NBT * 4);
    float*    addvec   = (float*)alloc((size_t)DIM * 4);
    float*    x2       = (float*)alloc((size_t)LTOK * DIM * 4);
    _Float16* xln16    = (_Float16*)alloc((size_t)LTOK * DIM * 2);
    _Float16* h16      = (_Float16*)alloc((size_t)LTOK * HID * 2);

    auto cvt = [&](const float* src, _Float16* dst, int n) {
        k_f2h<<<(n + 255) / 256, 256, 0, stream>>>(src, dst, n);
    };
    auto gemm = [&](const _Float16* A, const _Float16* W, const float* bias,
                    const float* res, float* oF, _Float16* oH,
                    int M, int N, int K, int mode) {
        dim3 g(N / 64, M / 64);
        k_gemm<<<g, 128, 0, stream>>>(A, W, bias, res, oF, oH, M, N, K, mode);
    };

    cvt(qkv_w, qkvW16, 3 * DIM * DIM);
    cvt(mproj_w, mprojW16, DIM * DIM);
    cvt(kv_w, kvW16, 2 * DIM * DIM);
    cvt(fc1_w, fc1W16, HID * DIM);
    cvt(fc2_w, fc2W16, DIM * HID);

    k_conv<<<(LTOK * DIM + 255) / 256, 256, 0, stream>>>(x, conv_w, conv_b, c);
    k_means<<<NP, 128, 0, stream>>>(c, means);
    k_rank<<<1, 256, 0, stream>>>(means, rankA, order);
    k_gather<<<(2 * MQP * DIM + 255) / 256, 256, 0, stream>>>(c, qa, order, xf16, xb16);

    // qkv: (1600x384) @ (1152x384)^T
    gemm(xf16, qkvW16, qkv_b, nullptr, qkv_out, nullptr, MQP, 3 * DIM, DIM, 0);
    k_qkvpost<<<(MQP * DIM + 255) / 256, 256, 0, stream>>>(qkv_out, q16, k16, vT16);

    k_attn<<<MQP / 16, 192, 0, stream>>>(q16, k16, vT16, ap1_w, ap1_b, ap2_w, ap2_b, attn16);

    // mproj: (1600x384) @ (384x384)^T
    gemm(attn16, mprojW16, mproj_b, nullptr, f_out, nullptr, MQP, DIM, DIM, 0);
    // kv: (1600x384, zero-padded) @ (768x384)^T
    gemm(xb16, kvW16, kv_b, nullptr, kv_out, nullptr, MQP, 2 * DIM, DIM, 0);

    k_sqa<<<1, 256, 0, stream>>>(kv_out, f_out, sp1_w, sp1_b, sp2_w, sp2_b,
                                 pn_w, pn_b, logits, addvec, out + (size_t)LTOK * DIM);

    k_scatter<<<LTOK, 128, 0, stream>>>(x, c, f_out, addvec, rankA,
                                        n1w, n1b, n2w, n2b, x2, xln16);

    // fc1: (3136x384) @ (1536x384)^T, silu -> f16
    gemm(xln16, fc1W16, fc1_b, nullptr, nullptr, h16, LTOK, HID, DIM, 2);
    // fc2: (3136x1536) @ (384x1536)^T + residual -> d_out
    gemm(h16, fc2W16, fc2_b, x2, out, nullptr, LTOK, DIM, HID, 3);
}
